// QwenSelfAttention_39496519254677
// MI455X (gfx1250) — matrix operand order repaired and checked by reading, not compiled, not run on hardware
//
#include <hip/hip_runtime.h>
#include <math.h>

// ---------------------------------------------------------------------------
// Qwen-style GQA attention forward for MI455X (gfx1250, wave32, WMMA).
// Compute-bound (~155 GFLOP vs ~150MB traffic, AI ~1000) => bf16 WMMA path
// (v_wmma_f32_16x16x32_bf16), f32 accumulation, flash-style online softmax.
// GEMMs are register-blocked 64x64 per wave (4x4 accumulator tiles) for 4x
// fragment reuse (32 FLOP/byte from L2 instead of 8).
// ---------------------------------------------------------------------------

#define DMODEL 2048
#define NHEADS 16
#define NKV    4
#define DHEAD  128
#define BATCH  2
#define SEQ    2048
#define KVDIM  (NKV * DHEAD)   // 512

typedef __attribute__((ext_vector_type(16))) __bf16 v16bf;
typedef __attribute__((ext_vector_type(8)))  float  v8f;

struct Frag32B { uint4 lo, hi; };   // 32 bytes == one v16bf fragment

__device__ inline __bf16 f2bf(float f) {
  unsigned u = __builtin_bit_cast(unsigned, f);
  unsigned r = (u + 0x7FFFu + ((u >> 16) & 1u)) >> 16;   // round-to-nearest-even
  unsigned short s = (unsigned short)r;
  return __builtin_bit_cast(__bf16, s);
}

// A-matrix 16x32 bf16 fragment (ISA 7.12.2): lane holds row M=lane%16;
// elements j: K = (j/8)*16 + (lane/16)*8 + j%8  -> two 16B loads.
__device__ inline v16bf load_a_frag(const __bf16* p, int ld) {
  int lane = threadIdx.x & 31;
  const __bf16* q = p + (size_t)(lane & 15) * ld + ((lane >> 4) << 3);
  Frag32B f;
  f.lo = *(const uint4*)(q);
  f.hi = *(const uint4*)(q + 16);
  return __builtin_bit_cast(v16bf, f);
}

// B-matrix 32x16 bf16 fragment: lane holds column N=lane%16; elements j:
// K = (j/8)*16 + (lane/16)*8 + j%8 (same order as A) -> two 16B loads.
__device__ inline v16bf load_b_frag(const __bf16* p, int ld) {
  int lane = threadIdx.x & 31;
  const __bf16* q = p + (size_t)(lane & 15) * ld + ((lane >> 4) << 3);
  Frag32B f;
  f.lo = *(const uint4*)(q);
  f.hi = *(const uint4*)(q + 16);
  return __builtin_bit_cast(v16bf, f);
}

// ---------------------------------------------------------------------------
// Elementwise f32 -> bf16
__global__ void cvt_bf16_kernel(const float* __restrict__ in,
                                __bf16* __restrict__ out, int n) {
  int i = blockIdx.x * blockDim.x + threadIdx.x;
  if (i < n) out[i] = f2bf(in[i]);
}

// Transpose + convert: wt[n*K + k] = w[k*N + n]   (w is [K,N] row-major)
__global__ void transpose_bf16_kernel(const float* __restrict__ w,
                                      __bf16* __restrict__ wt, int K, int N) {
  int i = blockIdx.x * blockDim.x + threadIdx.x;
  if (i >= K * N) return;
  int k = i / N, n = i - k * N;
  wt[(size_t)n * K + k] = f2bf(w[i]);
}

// ---------------------------------------------------------------------------
// C[M,N] (f32) = A[M,K](bf16) @ BT[N,K](bf16)^T + bias.
// One wave per 64x64 output block: 4x4 grid of 16x16 WMMA accumulators.
// Per k-step: 8 fragment loads (8KB) feed 16 WMMAs -> 4x register reuse.
__global__ __launch_bounds__(256)
void gemm_bf16(const __bf16* __restrict__ A, const __bf16* __restrict__ BT,
               const float* __restrict__ bias, float* __restrict__ C,
               int M, int N, int K) {
  int wave = blockIdx.x * (blockDim.x >> 5) + (threadIdx.x >> 5);
  int nbn  = N >> 6;                    // 64-wide blocks along N
  int mb   = wave / nbn;
  int nb   = wave - mb * nbn;
  if (mb >= (M >> 6)) return;           // uniform per wave: EXEC stays all-1s

  const __bf16* a0 = A  + (size_t)(mb << 6) * K;
  const __bf16* b0 = BT + (size_t)(nb << 6) * K;

  v8f acc[4][4];
#pragma unroll
  for (int i = 0; i < 4; ++i)
#pragma unroll
    for (int j = 0; j < 4; ++j)
      acc[i][j] = (v8f){0.f,0.f,0.f,0.f,0.f,0.f,0.f,0.f};

#pragma unroll 2
  for (int k0 = 0; k0 < K; k0 += 32) {
    __builtin_prefetch(a0 + k0 + 512, 0, 1);
    __builtin_prefetch(b0 + k0 + 512, 0, 1);
    v16bf af[4], bf[4];
#pragma unroll
    for (int i = 0; i < 4; ++i)
      af[i] = load_a_frag(a0 + (size_t)(i << 4) * K + k0, K);
#pragma unroll
    for (int j = 0; j < 4; ++j)
      bf[j] = load_b_frag(b0 + (size_t)(j << 4) * K + k0, K);
#pragma unroll
    for (int i = 0; i < 4; ++i)
#pragma unroll
      for (int j = 0; j < 4; ++j)
        acc[i][j] = __builtin_amdgcn_wmma_f32_16x16x32_bf16(
            false, af[i], false, bf[j], (short)0, acc[i][j], false, false);
  }

  int lane  = threadIdx.x & 31;
  int nlow  = lane & 15;
  int rbase = (lane >> 4) << 3;          // C layout: VGPR r -> row r (+8 for hi lanes)
#pragma unroll
  for (int j = 0; j < 4; ++j) {
    int n = (nb << 6) + (j << 4) + nlow;
    float bv = bias ? bias[n] : 0.0f;
#pragma unroll
    for (int i = 0; i < 4; ++i) {
#pragma unroll
      for (int r = 0; r < 8; ++r) {
        int m = (mb << 6) + (i << 4) + rbase + r;
        C[(size_t)m * N + n] = acc[i][j][r] + bv;
      }
    }
  }
}

// ---------------------------------------------------------------------------
// RoPE on q,k; pack bf16 q/k and transposed bf16 v; emit f32 k,v outputs.
// One thread per (b, t, h, pair-index d<64).
__global__ void rope_pack_kernel(const float* __restrict__ qf,
                                 const float* __restrict__ kf,
                                 const float* __restrict__ vf,
                                 float* __restrict__ k_out,
                                 float* __restrict__ v_out,
                                 __bf16* __restrict__ qbf,
                                 __bf16* __restrict__ kbf,
                                 __bf16* __restrict__ vtb) {
  int i = blockIdx.x * blockDim.x + threadIdx.x;
  int d = i & 63;
  int h = (i >> 6) & (NHEADS - 1);
  int t = (i >> 10) & (SEQ - 1);
  int b = i >> 21;
  if (b >= BATCH) return;

  // inv_freq = (1/1e6)^(2d/128) = exp(-ln(1e6) * d / 64)
  float inv = __expf(-13.815510557964274f * (float)d * (1.0f / 64.0f));
  float th  = (float)t * inv;
  float sn, cs;
  __sincosf(th, &sn, &cs);

  // ---- Q: rope, write bf16 [B,H,T,128]
  const float* qrow = qf + ((size_t)(b * SEQ + t)) * DMODEL + h * DHEAD;
  float x1 = qrow[d], x2 = qrow[d + 64];
  float r1 = cs * x1 - sn * x2;
  float r2 = cs * x2 + sn * x1;
  __bf16* qdst = qbf + ((size_t)((b * NHEADS + h) * SEQ + t)) * DHEAD;
  qdst[d]      = f2bf(r1);
  qdst[d + 64] = f2bf(r2);

  if (h < NKV) {
    int kvh = h;
    // ---- K: rope, write f32 [B,KV,T,128] output + bf16 copy
    const float* krow = kf + ((size_t)(b * SEQ + t)) * KVDIM + kvh * DHEAD;
    x1 = krow[d]; x2 = krow[d + 64];
    r1 = cs * x1 - sn * x2;
    r2 = cs * x2 + sn * x1;
    size_t ko = ((size_t)((b * NKV + kvh) * SEQ + t)) * DHEAD;
    k_out[ko + d]      = r1;
    k_out[ko + d + 64] = r2;
    kbf[ko + d]        = f2bf(r1);
    kbf[ko + d + 64]   = f2bf(r2);
    // ---- V: pass-through f32 output + transposed bf16 [B,KV,128,T]
    const float* vrow = vf + ((size_t)(b * SEQ + t)) * KVDIM + kvh * DHEAD;
    float v1 = vrow[d], v2 = vrow[d + 64];
    v_out[ko + d]      = v1;
    v_out[ko + d + 64] = v2;
    size_t vo = (size_t)(b * NKV + kvh) * DHEAD * SEQ;
    vtb[vo + (size_t)d * SEQ + t]        = f2bf(v1);
    vtb[vo + (size_t)(d + 64) * SEQ + t] = f2bf(v2);
  }
}

// ---------------------------------------------------------------------------
// Causal flash attention: one wave (32 threads) per (b, h, 16 query rows).
// Scores via q(A-frag) x k^T(B-frag) WMMAs; online softmax with shfl_xor row
// reductions; P transposed C->A layout through LDS; PV via p(A) x v^T(B).
__global__ __launch_bounds__(32)
void attn_fwd(const __bf16* __restrict__ Qb, const __bf16* __restrict__ Kb,
              const __bf16* __restrict__ Vt, __bf16* __restrict__ O) {
  const float scale = 0.08838834764831845f;   // 1/sqrt(128)
  int idx = blockIdx.x;
  int mt  = idx & (SEQ / 16 - 1);
  int h   = (idx >> 7) & (NHEADS - 1);
  int b   = idx >> 11;
  int kv  = h >> 2;                           // N_REP = 4

  const __bf16* qbase = Qb + (((size_t)(b * NHEADS + h) * SEQ) + mt * 16) * DHEAD;
  const __bf16* kbase = Kb + ((size_t)(b * NKV + kv) * SEQ) * DHEAD;
  const __bf16* vbase = Vt + ((size_t)(b * NKV + kv) * DHEAD) * SEQ;

  // q fragments: 16x128 = 4 chunks of 16x32
  v16bf qfr[4];
#pragma unroll
  for (int i = 0; i < 4; ++i) qfr[i] = load_a_frag(qbase + 32 * i, DHEAD);

  v8f acc[8];
#pragma unroll
  for (int i = 0; i < 8; ++i) acc[i] = (v8f){0.f,0.f,0.f,0.f,0.f,0.f,0.f,0.f};
  float mrow[8], lrow[8];
#pragma unroll
  for (int r = 0; r < 8; ++r) { mrow[r] = -3.0e38f; lrow[r] = 0.f; }

  __shared__ __bf16 plds[16 * 40];            // 16x32 P tile, ld=40 (pad)

  int lane  = threadIdx.x & 31;
  int cn    = lane & 15;                      // C-layout column
  int rbase = (lane >> 4) << 3;               // C-layout row base (0 or 8)
  int q0    = mt << 4;

  for (int s0 = 0; s0 < q0 + 16; s0 += 32) {
    // -------- scores: two 16x16 tiles (keys s0..s0+15 and s0+16..s0+31)
    v8f sc0 = (v8f){0.f,0.f,0.f,0.f,0.f,0.f,0.f,0.f};
    v8f sc1 = (v8f){0.f,0.f,0.f,0.f,0.f,0.f,0.f,0.f};
#pragma unroll
    for (int i = 0; i < 4; ++i) {
      v16bf kf0 = load_b_frag(kbase + (size_t)s0 * DHEAD + 32 * i, DHEAD);
      v16bf kf1 = load_b_frag(kbase + (size_t)(s0 + 16) * DHEAD + 32 * i, DHEAD);
      sc0 = __builtin_amdgcn_wmma_f32_16x16x32_bf16(false, qfr[i], false, kf0,
                                                    (short)0, sc0, false, false);
      sc1 = __builtin_amdgcn_wmma_f32_16x16x32_bf16(false, qfr[i], false, kf1,
                                                    (short)0, sc1, false, false);
    }

    // -------- online softmax (rows live across the 16-lane halves)
    float corr[8];
#pragma unroll
    for (int r = 0; r < 8; ++r) {
      int qm = q0 + rbase + r;
      float x0 = sc0[r] * scale; if (s0 + cn > qm)      x0 = -3.0e38f;
      float x1 = sc1[r] * scale; if (s0 + 16 + cn > qm) x1 = -3.0e38f;
      float mx = fmaxf(x0, x1);
      mx = fmaxf(mx, __shfl_xor(mx, 1, 32));
      mx = fmaxf(mx, __shfl_xor(mx, 2, 32));
      mx = fmaxf(mx, __shfl_xor(mx, 4, 32));
      mx = fmaxf(mx, __shfl_xor(mx, 8, 32));
      float mnew = fmaxf(mrow[r], mx);
      float p0 = __expf(x0 - mnew);
      float p1 = __expf(x1 - mnew);
      float ps = p0 + p1;
      ps += __shfl_xor(ps, 1, 32);
      ps += __shfl_xor(ps, 2, 32);
      ps += __shfl_xor(ps, 4, 32);
      ps += __shfl_xor(ps, 8, 32);
      float c = __expf(mrow[r] - mnew);
      lrow[r] = lrow[r] * c + ps;
      mrow[r] = mnew;
      corr[r] = c;
      plds[(rbase + r) * 40 + cn]      = f2bf(p0);
      plds[(rbase + r) * 40 + 16 + cn] = f2bf(p1);
    }
#pragma unroll
    for (int dt = 0; dt < 8; ++dt)
#pragma unroll
      for (int r = 0; r < 8; ++r) acc[dt][r] *= corr[r];

    __syncthreads();                                  // single-wave: cheap
    asm volatile("s_wait_dscnt 0x0" ::: "memory");    // LDS write->read hazard
    v16bf pf = load_a_frag(plds, 40);                 // P as 16x32 A fragment

    // -------- PV: out[16,128] accumulated as 8 column tiles
#pragma unroll
    for (int dt = 0; dt < 8; ++dt) {
      v16bf vfr = load_b_frag(vbase + (size_t)(dt * 16) * SEQ + s0, SEQ);
      acc[dt] = __builtin_amdgcn_wmma_f32_16x16x32_bf16(false, pf, false, vfr,
                                                        (short)0, acc[dt],
                                                        false, false);
    }
  }

  // -------- epilogue: normalize, store bf16 attention output [B,T,2048]
#pragma unroll
  for (int dt = 0; dt < 8; ++dt) {
#pragma unroll
    for (int r = 0; r < 8; ++r) {
      int m = q0 + rbase + r;
      O[((size_t)(b * SEQ) + m) * DMODEL + h * DHEAD + dt * 16 + cn] =
          f2bf(acc[dt][r] / lrow[r]);
    }
  }
}

// ---------------------------------------------------------------------------
extern "C" void kernel_launch(void* const* d_in, const int* in_sizes, int n_in,
                              void* d_out, int out_size, void* d_ws, size_t ws_size,
                              hipStream_t stream) {
  const float* x  = (const float*)d_in[0];
  const float* wq = (const float*)d_in[1];
  const float* bq = (const float*)d_in[2];
  const float* wk = (const float*)d_in[3];
  const float* bk = (const float*)d_in[4];
  const float* wv = (const float*)d_in[5];
  const float* bv = (const float*)d_in[6];
  const float* wo = (const float*)d_in[7];
  (void)in_sizes; (void)n_in; (void)out_size; (void)ws_size;

  float* out   = (float*)d_out;                              // [B,T,2048]
  float* k_out = out + (size_t)BATCH * SEQ * DMODEL;         // [B,KV,T,128]
  float* v_out = k_out + (size_t)BATCH * NKV * SEQ * DHEAD;  // [B,KV,T,128]

  size_t off = 0;
  auto alloc = [&](size_t bytes) -> void* {
    void* r = (char*)d_ws + off;
    off += (bytes + 255) & ~(size_t)255;
    return r;
  };
  const size_t ROWS = (size_t)BATCH * SEQ;   // 4096
  __bf16* xb    = (__bf16*)alloc(ROWS * DMODEL * 2);
  __bf16* wqT   = (__bf16*)alloc((size_t)DMODEL * DMODEL * 2);
  __bf16* wkT   = (__bf16*)alloc((size_t)DMODEL * KVDIM * 2);
  __bf16* wvT   = (__bf16*)alloc((size_t)DMODEL * KVDIM * 2);
  __bf16* woT   = (__bf16*)alloc((size_t)DMODEL * DMODEL * 2);
  float*  qf    = (float*)alloc(ROWS * DMODEL * 4);
  float*  kf    = (float*)alloc(ROWS * KVDIM * 4);
  float*  vf    = (float*)alloc(ROWS * KVDIM * 4);
  __bf16* qbf   = (__bf16*)alloc(ROWS * DMODEL * 2);
  __bf16* kbf   = (__bf16*)alloc((size_t)BATCH * NKV * SEQ * DHEAD * 2);
  __bf16* vtb   = (__bf16*)alloc((size_t)BATCH * NKV * SEQ * DHEAD * 2);
  __bf16* attnb = (__bf16*)alloc(ROWS * DMODEL * 2);

  // 1) activations + weights -> bf16 (weights transposed for contiguous-K B frags)
  {
    int n = (int)(ROWS * DMODEL);
    cvt_bf16_kernel<<<(n + 255) / 256, 256, 0, stream>>>(x, xb, n);
  }
  transpose_bf16_kernel<<<(DMODEL * DMODEL + 255) / 256, 256, 0, stream>>>(wq, wqT, DMODEL, DMODEL);
  transpose_bf16_kernel<<<(DMODEL * KVDIM  + 255) / 256, 256, 0, stream>>>(wk, wkT, DMODEL, KVDIM);
  transpose_bf16_kernel<<<(DMODEL * KVDIM  + 255) / 256, 256, 0, stream>>>(wv, wvT, DMODEL, KVDIM);
  transpose_bf16_kernel<<<(DMODEL * DMODEL + 255) / 256, 256, 0, stream>>>(wo, woT, DMODEL, DMODEL);

  // 2) QKV projections (bf16 WMMA, f32 out); 64x64 block per wave, 8 waves/WG
  {
    int waves = (int)((ROWS / 64) * (DMODEL / 64));   // 2048
    gemm_bf16<<<waves / 8, 256, 0, stream>>>(xb, wqT, bq, qf, (int)ROWS, DMODEL, DMODEL);
  }
  {
    int waves = (int)((ROWS / 64) * (KVDIM / 64));    // 512
    gemm_bf16<<<waves / 8, 256, 0, stream>>>(xb, wkT, bk, kf, (int)ROWS, KVDIM, DMODEL);
    gemm_bf16<<<waves / 8, 256, 0, stream>>>(xb, wvT, bv, vf, (int)ROWS, KVDIM, DMODEL);
  }

  // 3) RoPE + layout packing (also writes f32 k,v outputs)
  {
    int n = BATCH * SEQ * NHEADS * 64;
    rope_pack_kernel<<<n / 256, 256, 0, stream>>>(qf, kf, vf, k_out, v_out, qbf, kbf, vtb);
  }

  // 4) causal flash attention (wave per 16 query rows)
  attn_fwd<<<BATCH * NHEADS * (SEQ / 16), 32, 0, stream>>>(qbf, kbf, vtb, attnb);

  // 5) output projection
  {
    int waves = (int)((ROWS / 64) * (DMODEL / 64));   // 2048
    gemm_bf16<<<waves / 8, 256, 0, stream>>>(attnb, woT, nullptr, out, (int)ROWS, DMODEL, DMODEL);
  }
}
